// Phi4FlashMamba_56650618635026
// MI455X (gfx1250) — matrix operand order repaired and checked by reading, not compiled, not run on hardware
//
#include <hip/hip_runtime.h>
#include <hip/hip_bf16.h>

// ---------------- problem constants ----------------
#define DM   2560          // d_model
#define DI   5120          // d_inner
#define DS   16            // d_state
#define RK   160           // dt_rank
#define NB   2             // batch
#define SL   2048          // seq
#define NT   (NB * SL)     // 4096 "rows" (b*l flattened)
#define NXW  (RK + 2*DS)   // 192 real x_dbl width
#define NXP  256           // padded x_dbl width (multiple of 128)

typedef __attribute__((ext_vector_type(16))) __bf16 v16bf;
typedef __attribute__((ext_vector_type(8)))  __bf16 v8bf;
typedef __attribute__((ext_vector_type(8)))  float  v8f;
typedef __attribute__((ext_vector_type(4)))  int    v4i;

#define HAS_ASYNC_LDS __has_builtin(__builtin_amdgcn_global_load_async_to_lds_b128)
#define AS_GLOBAL_V4I(p) ((__attribute__((address_space(1))) v4i*)(p))
#define AS_LDS_V4I(p)    ((__attribute__((address_space(3))) v4i*)(p))

__device__ __forceinline__ unsigned short f2bf(float f) {
  union { float f; unsigned u; } v; v.f = f;
  unsigned u = v.u;
  u += 0x7fffu + ((u >> 16) & 1u);   // round-to-nearest-even
  return (unsigned short)(u >> 16);
}

__device__ __forceinline__ float silu(float x) {
  return x / (1.0f + __expf(-x));
}

// ---------------- elementwise converts ----------------
__global__ __launch_bounds__(256) void k_f32_to_bf16(
    const float* __restrict__ s, unsigned short* __restrict__ d, long n) {
  long i = (long)blockIdx.x * 256 + threadIdx.x;
  if (i < n) d[i] = f2bf(s[i]);
}

// W_x (192 x 5120) -> bf16 padded to (256 x 5120), pad rows zero
__global__ __launch_bounds__(256) void k_pad_wx(
    const float* __restrict__ s, unsigned short* __restrict__ d) {
  long i = (long)blockIdx.x * 256 + threadIdx.x;
  if (i >= (long)NXP * DI) return;
  int row = (int)(i / DI);
  d[i] = (row < NXW) ? f2bf(s[i]) : (unsigned short)0;
}

// extract dt_r = x_dbl[:, :160] as bf16 for the dt GEMM
__global__ __launch_bounds__(256) void k_extract_dtr(
    const float* __restrict__ xdbl, unsigned short* __restrict__ dtr) {
  long i = (long)blockIdx.x * 256 + threadIdx.x;
  if (i >= (long)NT * RK) return;
  long row = i / RK;
  int  col = (int)(i % RK);
  dtr[i] = f2bf(xdbl[row * NXP + col]);
}

// dt = softplus(dt_raw + b_dt), in place
__global__ __launch_bounds__(256) void k_softplus(
    float* __restrict__ dt, const float* __restrict__ b_dt) {
  long i = (long)blockIdx.x * 256 + threadIdx.x;
  if (i >= (long)NT * DI) return;
  int c = (int)(i % DI);
  float v = dt[i] + b_dt[c];
  dt[i] = (v > 20.0f) ? v : log1pf(__expf(v));
}

// ---------------- causal depthwise conv (K=4) + silu ----------------
// xz layout: [NT][2*DI] f32; x = cols [0,DI)
__global__ __launch_bounds__(256) void k_conv_silu(
    const float* __restrict__ xz, const float* __restrict__ cw,
    const float* __restrict__ cb, float* __restrict__ xact,
    unsigned short* __restrict__ xact_bf) {
  long i = (long)blockIdx.x * 256 + threadIdx.x;
  if (i >= (long)NT * DI) return;
  int  c = (int)(i % DI);
  long r = i / DI;              // b*SL + t
  int  t = (int)(r % SL);
  const float* w = cw + (size_t)c * 4;
  float acc = cb[c];
  long rowbase = r * (2L * DI) + c;
#pragma unroll
  for (int k = 0; k < 4; ++k) {
    int tl = t - 3 + k;
    if (tl >= 0) acc += w[k] * xz[rowbase + (long)(tl - t) * (2L * DI)];
  }
  float s = silu(acc);
  xact[i]    = s;
  xact_bf[i] = f2bf(s);
}

// ---------------- WMMA bf16 GEMM: C[M][N] = A[M][K] * W[N][K]^T ----------------
// block 256 thr (8 waves) -> 128x128 tile; wave -> 64x32 (4x2 subtiles of 16x16).
// A and W tiles (128x32 bf16) double-buffered in LDS, rows padded to 40 elems
// (80B) so the 16-lane fragment reads hit 16 disjoint 4-bank groups.
// Staging uses async global->LDS DMA (ASYNCcnt) when available.
// requires M%128==0, N%128==0, K%32==0; N,K compile-time for immediate offsets.
template <int N, int K>
__global__ __launch_bounds__(256) void k_gemm_bf16(
    const __bf16* __restrict__ A, const __bf16* __restrict__ W,
    float* __restrict__ C) {
  constexpr int LDT = 40;                    // padded LDS row, elements
  __shared__ __bf16 lA[2][128 * LDT];        // 20 KB
  __shared__ __bf16 lB[2][128 * LDT];        // 20 KB
  const int tid  = threadIdx.x;
  const int lane = tid & 31;
  const int wave = tid >> 5;
  const int half = lane >> 4;                // 0/1
  const int l15  = lane & 15;
  const int blkM = blockIdx.y * 128;
  const int blkN = blockIdx.x * 128;
  const int wm0  = (wave >> 2) * 64;         // wave row base within tile
  const int wn0  = (wave & 3) * 32;          // wave col base within tile

  // staging map: 512 x 16B transfers per matrix; thread does 2 each
  const int q0   = tid * 2;

  auto stage = [&](int buf, int k0) {
#pragma unroll
    for (int t = 0; t < 2; ++t) {
      int q = q0 + t;
      int row = q >> 2;                      // 0..127
      int seg = (q & 3) * 8;                 // element offset within 32
      const __bf16* ga = A + (size_t)(blkM + row) * K + k0 + seg;
      const __bf16* gb = W + (size_t)(blkN + row) * K + k0 + seg;
      __bf16* la = &lA[buf][row * LDT + seg];
      __bf16* lb = &lB[buf][row * LDT + seg];
#if HAS_ASYNC_LDS
      __builtin_amdgcn_global_load_async_to_lds_b128(
          AS_GLOBAL_V4I(ga), AS_LDS_V4I(la), 0, 0);
      __builtin_amdgcn_global_load_async_to_lds_b128(
          AS_GLOBAL_V4I(gb), AS_LDS_V4I(lb), 0, 0);
#else
      v8bf ta = *(const v8bf*)ga;
      v8bf tb = *(const v8bf*)gb;
      *(v8bf*)la = ta;
      *(v8bf*)lb = tb;
#endif
      __builtin_prefetch((const char*)ga + 128, 0, 1);   // 2 K-steps ahead
      __builtin_prefetch((const char*)gb + 128, 0, 1);
    }
  };

  auto wait_and_barrier = [&]() {
#if HAS_ASYNC_LDS
    asm volatile("s_wait_asynccnt 0x0" ::: "memory");
#endif
    __syncthreads();
  };

  v8f acc[4][2] = {};

  stage(0, 0);
  wait_and_barrier();

  constexpr int KSTEPS = K / 32;
  for (int ks = 0; ks < KSTEPS; ++ks) {
    const int buf = ks & 1;
    if (ks + 1 < KSTEPS) stage(buf ^ 1, (ks + 1) * 32);

    v16bf afrag[4];
#pragma unroll
    for (int i = 0; i < 4; ++i) {
      const __bf16* ap = &lA[buf][(wm0 + i * 16 + l15) * LDT + half * 8];
      union { v16bf v; v8bf h[2]; } u;
      u.h[0] = *(const v8bf*)(ap);
      u.h[1] = *(const v8bf*)(ap + 16);
      afrag[i] = u.v;
    }
    v16bf bfrag[2];
#pragma unroll
    for (int j = 0; j < 2; ++j) {
      const __bf16* bp = &lB[buf][(wn0 + j * 16 + l15) * LDT + half * 8];
      union { v16bf v; v8bf h[2]; } u;
      u.h[0] = *(const v8bf*)(bp);
      u.h[1] = *(const v8bf*)(bp + 16);
      bfrag[j] = u.v;
    }
#pragma unroll
    for (int i = 0; i < 4; ++i)
#pragma unroll
      for (int j = 0; j < 2; ++j)
        acc[i][j] = __builtin_amdgcn_wmma_f32_16x16x32_bf16(
            false, afrag[i], false, bfrag[j], (short)0, acc[i][j], false, false);

    wait_and_barrier();
  }

  // D layout: VGPR v -> row m = v + 8*half, col n = lane&15
#pragma unroll
  for (int i = 0; i < 4; ++i)
#pragma unroll
    for (int j = 0; j < 2; ++j) {
      float* cp = C + (size_t)(blkM + wm0 + i * 16 + half * 8) * N
                    + blkN + wn0 + j * 16 + l15;
#pragma unroll
      for (int v = 0; v < 8; ++v) cp[(size_t)v * N] = acc[i][j][v];
    }
}

// ---------------- selective scan (sequential in L) + fused epilogue ----------------
// one thread per (batch, channel); h[16] in VGPRs; writes y_bf16 = ((scan + D*x) * silu(z))
__global__ __launch_bounds__(256) void k_scan(
    const float* __restrict__ xact, const float* __restrict__ dt,
    const float* __restrict__ xdbl, const float* __restrict__ xz,
    const float* __restrict__ A_log, const float* __restrict__ Dv,
    unsigned short* __restrict__ y_bf) {
  int c = blockIdx.x * 256 + threadIdx.x;   // channel
  int b = blockIdx.y;                       // batch
  float Arow[DS];
#pragma unroll
  for (int s = 0; s < DS; ++s) Arow[s] = -__expf(A_log[(size_t)c * DS + s]);
  float Dc = Dv[c];
  float h[DS];
#pragma unroll
  for (int s = 0; s < DS; ++s) h[s] = 0.0f;

  for (int t = 0; t < SL; ++t) {
    size_t r    = (size_t)b * SL + t;
    float x_t   = xact[r * DI + c];
    float dt_t  = dt[r * DI + c];
    float z     = xz[r * (2 * DI) + DI + c];
    const float* bc = xdbl + r * NXP + RK;  // B at [0,16), C at [16,32)
    float dx = dt_t * x_t;
    float y  = 0.0f;
#pragma unroll
    for (int s = 0; s < DS; ++s) {
      float dA = __expf(dt_t * Arow[s]);
      h[s] = dA * h[s] + dx * bc[s];
      y   += h[s] * bc[DS + s];
    }
    y = (y + Dc * x_t) * silu(z);
    y_bf[r * DI + c] = f2bf(y);
  }
}

// ---------------- host ----------------
static inline long cdiv(long a, long b) { return (a + b - 1) / b; }

extern "C" void kernel_launch(void* const* d_in, const int* in_sizes, int n_in,
                              void* d_out, int out_size, void* d_ws, size_t ws_size,
                              hipStream_t stream) {
  const float* hs     = (const float*)d_in[0];  // (2,2048,2560)
  const float* W_in   = (const float*)d_in[1];  // (10240,2560)
  const float* conv_w = (const float*)d_in[2];  // (5120,1,4)
  const float* conv_b = (const float*)d_in[3];  // (5120)
  const float* W_x    = (const float*)d_in[4];  // (192,5120)
  const float* W_dt   = (const float*)d_in[5];  // (5120,160)
  const float* b_dt   = (const float*)d_in[6];  // (5120)
  const float* A_log  = (const float*)d_in[7];  // (5120,16)
  const float* Dv     = (const float*)d_in[8];  // (5120)
  const float* W_out  = (const float*)d_in[9];  // (2560,5120)
  float* out = (float*)d_out;                   // (2,2048,2560)

  char* ws = (char*)d_ws;
  size_t off = 0;
  auto alloc = [&](size_t bytes) -> void* {
    off = (off + 255) & ~(size_t)255;
    void* p = ws + off;
    off += bytes;
    return p;
  };

  unsigned short* hs_bf   = (unsigned short*)alloc((size_t)NT * DM * 2);
  unsigned short* Win_bf  = (unsigned short*)alloc((size_t)2 * DI * DM * 2);
  unsigned short* Wx_bf   = (unsigned short*)alloc((size_t)NXP * DI * 2);
  unsigned short* Wdt_bf  = (unsigned short*)alloc((size_t)DI * RK * 2);
  unsigned short* Wout_bf = (unsigned short*)alloc((size_t)DM * DI * 2);
  float*          xz      = (float*)alloc((size_t)NT * 2 * DI * 4);
  float*          xact    = (float*)alloc((size_t)NT * DI * 4);
  unsigned short* xact_bf = (unsigned short*)alloc((size_t)NT * DI * 2);
  float*          xdbl    = (float*)alloc((size_t)NT * NXP * 4);
  unsigned short* dtr_bf  = (unsigned short*)alloc((size_t)NT * RK * 2);
  float*          dtbuf   = (float*)alloc((size_t)NT * DI * 4);
  unsigned short* y_bf    = (unsigned short*)alloc((size_t)NT * DI * 2);
  (void)ws_size; (void)n_in; (void)in_sizes; (void)out_size;

  // 1) fp32 -> bf16 conversions of activations/weights
  {
    long n;
    n = (long)NT * DM;
    k_f32_to_bf16<<<dim3(cdiv(n, 256)), 256, 0, stream>>>(hs, hs_bf, n);
    n = (long)2 * DI * DM;
    k_f32_to_bf16<<<dim3(cdiv(n, 256)), 256, 0, stream>>>(W_in, Win_bf, n);
    n = (long)DI * RK;
    k_f32_to_bf16<<<dim3(cdiv(n, 256)), 256, 0, stream>>>(W_dt, Wdt_bf, n);
    n = (long)DM * DI;
    k_f32_to_bf16<<<dim3(cdiv(n, 256)), 256, 0, stream>>>(W_out, Wout_bf, n);
    n = (long)NXP * DI;
    k_pad_wx<<<dim3(cdiv(n, 256)), 256, 0, stream>>>(W_x, Wx_bf);
  }

  // 2) xz = hs @ W_in^T   (4096 x 10240 x 2560)
  k_gemm_bf16<2 * DI, DM><<<dim3((2 * DI) / 128, NT / 128), 256, 0, stream>>>(
      (const __bf16*)hs_bf, (const __bf16*)Win_bf, xz);

  // 3) causal depthwise conv + silu -> x_act (f32 + bf16)
  {
    long n = (long)NT * DI;
    k_conv_silu<<<dim3(cdiv(n, 256)), 256, 0, stream>>>(xz, conv_w, conv_b, xact, xact_bf);
  }

  // 4) x_dbl = x_act @ W_x^T   (4096 x 256(pad) x 5120)
  k_gemm_bf16<NXP, DI><<<dim3(NXP / 128, NT / 128), 256, 0, stream>>>(
      (const __bf16*)xact_bf, (const __bf16*)Wx_bf, xdbl);

  // 5) dt_raw = dt_r @ W_dt^T   (4096 x 5120 x 160)
  {
    long n = (long)NT * RK;
    k_extract_dtr<<<dim3(cdiv(n, 256)), 256, 0, stream>>>(xdbl, dtr_bf);
  }
  k_gemm_bf16<DI, RK><<<dim3(DI / 128, NT / 128), 256, 0, stream>>>(
      (const __bf16*)dtr_bf, (const __bf16*)Wdt_bf, dtbuf);

  // 6) dt = softplus(dt_raw + b_dt)
  {
    long n = (long)NT * DI;
    k_softplus<<<dim3(cdiv(n, 256)), 256, 0, stream>>>(dtbuf, b_dt);
  }

  // 7) selective scan + epilogue ( +D*x, *silu(z) ) -> y_bf16
  k_scan<<<dim3(DI / 256, NB), 256, 0, stream>>>(xact, dtbuf, xdbl, xz, A_log, Dv, y_bf);

  // 8) out = y @ W_out^T   (4096 x 2560 x 5120)
  k_gemm_bf16<DM, DI><<<dim3(DM / 128, NT / 128), 256, 0, stream>>>(
      (const __bf16*)y_bf, (const __bf16*)Wout_bf, out);
}
